// CausalSelfAttention_22265110463090
// MI455X (gfx1250) — matrix operand order repaired and checked by reading, not compiled, not run on hardware
//
#include <hip/hip_runtime.h>

typedef __attribute__((ext_vector_type(16))) __bf16        v16bf;
typedef __attribute__((ext_vector_type(8)))  float         v8f;
typedef __attribute__((ext_vector_type(4)))  float         f32x4;
typedef __attribute__((ext_vector_type(4)))  unsigned int  u32x4;

#define T_SEQ 2048
#define C_DIM 768
#define NHEAD 12
#define HDIM  64

union FragU { v16bf v; u32x4 q[2]; };

// Native f32 -> bf16 (RNE) conversion; lowers to v_cvt*bf16_f32 on gfx1250.
__device__ __forceinline__ unsigned short f2bf(float f) {
  return __builtin_bit_cast(unsigned short, (__bf16)f);
}

__device__ __forceinline__ v8f wmma_bf16(v16bf a, v16bf b, v8f c) {
  return __builtin_amdgcn_wmma_f32_16x16x32_bf16(false, a, false, b, (short)0, c,
                                                 false, false);
}

// A fragment (16x32 bf16) from [m][k] storage, ld multiple of 8, 16B-aligned rows.
// lane: m = lane&15, k = k0 + (lane>>4)*8 + {0..7, 16..23}  -> 2x b128 loads.
__device__ __forceinline__ v16bf frag_a_ld(const unsigned short* p, int ld, int k0) {
  int lane = threadIdx.x & 31;
  const unsigned short* r = p + (lane & 15) * ld + k0 + ((lane >> 4) << 3);
  FragU u;
  u.q[0] = *(const u32x4*)(r);
  u.q[1] = *(const u32x4*)(r + 16);
  return u.v;
}

// B fragment (32x16 bf16) from [n][k] storage (transposed), ld multiple of 8.
// lane: n = n0 + (lane&15), k = k0 + (lane>>4)*8 + {0..7, 16..23}  -> 2x b128 loads.
__device__ __forceinline__ v16bf frag_bt_ld(const unsigned short* p, int ld, int n0, int k0) {
  int lane = threadIdx.x & 31;
  const unsigned short* r = p + (n0 + (lane & 15)) * ld + k0 + ((lane >> 4) << 3);
  FragU u;
  u.q[0] = *(const u32x4*)(r);
  u.q[1] = *(const u32x4*)(r + 16);
  return u.v;
}

// ---------------- Kernel 1: qkv = x @ w_qkv + b, fused RoPE, per-head bf16 out ----
__global__ __launch_bounds__(128)
void qkv_rope_kernel(const float* __restrict__ x, const float* __restrict__ w,
                     const float* __restrict__ bias,
                     unsigned short* __restrict__ qw, unsigned short* __restrict__ kw,
                     unsigned short* __restrict__ vw) {
  __shared__ __align__(16) unsigned short As[64 * 40];    // [m][k]
  __shared__ __align__(16) unsigned short Bst[64 * 40];   // [n][k] (transposed)
  int tid = threadIdx.x;
  int lane = tid & 31, wm = tid >> 5;
  int n0 = blockIdx.x * 64;     // column base in [0, 2304)
  int row0 = blockIdx.y * 64;   // sequence base

  v8f acc[4] = {{}, {}, {}, {}};
  for (int kk = 0; kk < C_DIM; kk += 32) {
    // stage A: 64 rows x 32 k, vectorized f32x4 loads + packed bf16 stores
#pragma unroll
    for (int it = 0; it < 4; ++it) {
      int idx = tid + it * 128;
      int r = idx >> 3, c4 = (idx & 7) << 2;
      f32x4 xv = *(const f32x4*)(x + (size_t)(row0 + r) * C_DIM + kk + c4);
      unsigned int lo = (unsigned int)f2bf(xv[0]) | ((unsigned int)f2bf(xv[1]) << 16);
      unsigned int hi = (unsigned int)f2bf(xv[2]) | ((unsigned int)f2bf(xv[3]) << 16);
      *(unsigned int*)(As + r * 40 + c4)     = lo;
      *(unsigned int*)(As + r * 40 + c4 + 2) = hi;
    }
    // stage B transposed: w is [k][n]; write Bst[n][k]
#pragma unroll
    for (int it = 0; it < 4; ++it) {
      int idx = tid + it * 128;
      int r = idx >> 4, c4 = (idx & 15) << 2;
      f32x4 wv = *(const f32x4*)(w + (size_t)(kk + r) * (3 * C_DIM) + n0 + c4);
      Bst[(c4 + 0) * 40 + r] = f2bf(wv[0]);
      Bst[(c4 + 1) * 40 + r] = f2bf(wv[1]);
      Bst[(c4 + 2) * 40 + r] = f2bf(wv[2]);
      Bst[(c4 + 3) * 40 + r] = f2bf(wv[3]);
    }
    if (kk + 32 < C_DIM) {
      __builtin_prefetch(x + (size_t)(row0 + (tid >> 1)) * C_DIM + kk + 32 + (tid & 1) * 16, 0, 0);
      __builtin_prefetch(w + (size_t)(kk + 32 + (tid >> 2)) * (3 * C_DIM) + n0 + (tid & 3) * 16, 0, 0);
    }
    __syncthreads();
    v16bf a = frag_a_ld(As + wm * 16 * 40, 40, 0);
#pragma unroll
    for (int j = 0; j < 4; ++j) {
      v16bf b = frag_bt_ld(Bst, 40, j * 16, 0);
      acc[j] = wmma_bf16(a, b, acc[j]);
    }
    __syncthreads();
  }

  int nl = lane & 15, hs = lane >> 4;
  int region = n0 / C_DIM;                 // 0=q 1=k 2=v
  int head = (n0 % C_DIM) / HDIM;
  unsigned short* dst = (region == 0) ? qw : ((region == 1) ? kw : vw);
  unsigned short* base = dst + head * (T_SEQ * HDIM);

  if (region < 2) {
    // RoPE: pair (d, d+32) -> accumulators j and j+2, same lane/element slot.
#pragma unroll
    for (int j = 0; j < 2; ++j) {
      int d = j * 16 + nl;                 // 0..31
      float b1 = bias[n0 + d];
      float b2 = bias[n0 + d + 32];
      float invf = __powf(10000.0f, -(float)d * (1.0f / 32.0f));
#pragma unroll
      for (int i = 0; i < 8; ++i) {
        int t = row0 + wm * 16 + i + hs * 8;
        float sn, cs;
        __sincosf((float)t * invf, &sn, &cs);
        float x1 = acc[j][i] + b1;
        float x2 = acc[j + 2][i] + b2;
        base[t * HDIM + d]      = f2bf(x1 * cs - x2 * sn);
        base[t * HDIM + d + 32] = f2bf(x2 * cs + x1 * sn);
      }
    }
  } else {
#pragma unroll
    for (int j = 0; j < 4; ++j) {
      int d = j * 16 + nl;
      float b1 = bias[n0 + d];
#pragma unroll
      for (int i = 0; i < 8; ++i) {
        int t = row0 + wm * 16 + i + hs * 8;
        base[t * HDIM + d] = f2bf(acc[j][i] + b1);
      }
    }
  }
}

// ---------------- Kernel 2: flash attention, one block per (head, 64-query tile) --
__global__ __launch_bounds__(128)
void attn_kernel(const unsigned short* __restrict__ qw, const unsigned short* __restrict__ kw,
                 const unsigned short* __restrict__ vw, unsigned short* __restrict__ yw) {
  __shared__ __align__(16) unsigned short Ks[32 * 72];        // [key][dim]
  __shared__ __align__(16) unsigned short Vst[64 * 40];       // [dim][key] (transposed)
  __shared__ __align__(16) unsigned short Ps[4 * 16 * 40];    // per-wave P tiles [m][k]
  int tid = threadIdx.x;
  int lane = tid & 31, wm = tid >> 5;
  int nl = lane & 15, hs = lane >> 4;
  int head = blockIdx.y;
  int q0 = blockIdx.x * 64 + wm * 16;
  const unsigned short* qh = qw + head * (T_SEQ * HDIM);
  const unsigned short* kh = kw + head * (T_SEQ * HDIM);
  const unsigned short* vh = vw + head * (T_SEQ * HDIM);
  unsigned short* Pw = Ps + wm * (16 * 40);

  // Q fragments straight from global: 2x global_load_b128 each
  v16bf qa0 = frag_a_ld(qh + q0 * HDIM, HDIM, 0);
  v16bf qa1 = frag_a_ld(qh + q0 * HDIM, HDIM, 32);

  v8f o[4] = {{}, {}, {}, {}};
  float mrun[8], lrun[8];
#pragma unroll
  for (int i = 0; i < 8; ++i) { mrun[i] = -1e30f; lrun[i] = 0.0f; }

  int kend = blockIdx.x * 64 + 64;
  for (int key0 = 0; key0 < kend; key0 += 32) {
    // K tile: row-wise b128 copy (32 rows x 8-ushort chunks)
#pragma unroll
    for (int it = 0; it < 2; ++it) {
      int idx = tid + it * 128;
      int r = idx >> 3, c8 = (idx & 7) << 3;
      *(u32x4*)(Ks + r * 72 + c8) = *(const u32x4*)(kh + (key0 + r) * HDIM + c8);
    }
    // V tile transposed: read b128 along dim, scatter u16 to [dim][key]
#pragma unroll
    for (int it = 0; it < 2; ++it) {
      int idx = tid + it * 128;
      int r = idx >> 3, c8 = (idx & 7) << 3;
      u32x4 d = *(const u32x4*)(vh + (key0 + r) * HDIM + c8);
#pragma unroll
      for (int e = 0; e < 4; ++e) {
        Vst[(c8 + 2 * e)     * 40 + r] = (unsigned short)(d[e] & 0xFFFFu);
        Vst[(c8 + 2 * e + 1) * 40 + r] = (unsigned short)(d[e] >> 16);
      }
    }
    __syncthreads();

    v8f s0 = {}, s1 = {};
    {
      v16bf b;
      b = frag_bt_ld(Ks, 72, 0, 0);    s0 = wmma_bf16(qa0, b, s0);
      b = frag_bt_ld(Ks, 72, 0, 32);   s0 = wmma_bf16(qa1, b, s0);
      b = frag_bt_ld(Ks, 72, 16, 0);   s1 = wmma_bf16(qa0, b, s1);
      b = frag_bt_ld(Ks, 72, 16, 32);  s1 = wmma_bf16(qa1, b, s1);
    }

    float sv0[8], sv1[8];
#pragma unroll
    for (int i = 0; i < 8; ++i) {
      int qi = q0 + i + hs * 8;
      sv0[i] = ((key0 + nl)      <= qi) ? s0[i] * 0.125f : -3.0e38f;
      sv1[i] = ((key0 + 16 + nl) <= qi) ? s1[i] * 0.125f : -3.0e38f;
    }
#pragma unroll
    for (int i = 0; i < 8; ++i) {
      float mx = fmaxf(sv0[i], sv1[i]);
#pragma unroll
      for (int off = 1; off < 16; off <<= 1)
        mx = fmaxf(mx, __shfl_xor(mx, off, 32));     // half-wave row reduce
      float m2 = fmaxf(mrun[i], mx);
      float resc = __expf(mrun[i] - m2);
      mrun[i] = m2;
      float p0 = __expf(sv0[i] - m2);
      float p1 = __expf(sv1[i] - m2);
      int m = i + hs * 8;
      Pw[m * 40 + nl]      = f2bf(p0);
      Pw[m * 40 + 16 + nl] = f2bf(p1);
      float ps = p0 + p1;
#pragma unroll
      for (int off = 1; off < 16; off <<= 1)
        ps += __shfl_xor(ps, off, 32);
      lrun[i] = lrun[i] * resc + ps;
#pragma unroll
      for (int j = 0; j < 4; ++j) o[j][i] *= resc;
    }
    __syncthreads();   // P C-layout -> A-layout restripe through LDS

    v16bf pa = frag_a_ld(Pw, 40, 0);
#pragma unroll
    for (int j = 0; j < 4; ++j) {
      v16bf vb = frag_bt_ld(Vst, 40, j * 16, 0);     // n=dim, k=key
      o[j] = wmma_bf16(pa, vb, o[j]);
    }
    __syncthreads();
  }

#pragma unroll
  for (int j = 0; j < 4; ++j)
#pragma unroll
    for (int i = 0; i < 8; ++i) {
      int t = q0 + i + hs * 8;
      yw[t * C_DIM + head * HDIM + j * 16 + nl] = f2bf(o[j][i] / lrun[i]);
    }
}

// ---------------- Kernel 3: out = y @ w_proj + b_proj (f32 out) --------------------
__global__ __launch_bounds__(128)
void proj_kernel(const unsigned short* __restrict__ yw, const float* __restrict__ w,
                 const float* __restrict__ bias, float* __restrict__ out) {
  __shared__ __align__(16) unsigned short As[64 * 40];    // [m][k]
  __shared__ __align__(16) unsigned short Bst[64 * 40];   // [n][k]
  int tid = threadIdx.x;
  int lane = tid & 31, wm = tid >> 5;
  int n0 = blockIdx.x * 64;
  int row0 = blockIdx.y * 64;
  v8f acc[4] = {{}, {}, {}, {}};
  for (int kk = 0; kk < C_DIM; kk += 32) {
    // A tile: already bf16, pure b128 row copies
#pragma unroll
    for (int it = 0; it < 2; ++it) {
      int idx = tid + it * 128;
      int r = idx >> 2, c8 = (idx & 3) << 3;
      *(u32x4*)(As + r * 40 + c8) =
          *(const u32x4*)(yw + (size_t)(row0 + r) * C_DIM + kk + c8);
    }
#pragma unroll
    for (int it = 0; it < 4; ++it) {
      int idx = tid + it * 128;
      int r = idx >> 4, c4 = (idx & 15) << 2;
      f32x4 wv = *(const f32x4*)(w + (size_t)(kk + r) * C_DIM + n0 + c4);
      Bst[(c4 + 0) * 40 + r] = f2bf(wv[0]);
      Bst[(c4 + 1) * 40 + r] = f2bf(wv[1]);
      Bst[(c4 + 2) * 40 + r] = f2bf(wv[2]);
      Bst[(c4 + 3) * 40 + r] = f2bf(wv[3]);
    }
    if (kk + 32 < C_DIM) {
      __builtin_prefetch(w + (size_t)(kk + 32 + (tid >> 2)) * C_DIM + n0 + (tid & 3) * 16, 0, 0);
    }
    __syncthreads();
    v16bf a = frag_a_ld(As + wm * 16 * 40, 40, 0);
#pragma unroll
    for (int j = 0; j < 4; ++j) {
      v16bf b = frag_bt_ld(Bst, 40, j * 16, 0);
      acc[j] = wmma_bf16(a, b, acc[j]);
    }
    __syncthreads();
  }
  int nl = lane & 15, hs = lane >> 4;
#pragma unroll
  for (int j = 0; j < 4; ++j) {
    int col = n0 + j * 16 + nl;
    float b1 = bias[col];
#pragma unroll
    for (int i = 0; i < 8; ++i) {
      int t = row0 + wm * 16 + i + hs * 8;
      out[t * C_DIM + col] = acc[j][i] + b1;
    }
  }
}

extern "C" void kernel_launch(void* const* d_in, const int* in_sizes, int n_in,
                              void* d_out, int out_size, void* d_ws, size_t ws_size,
                              hipStream_t stream) {
  (void)in_sizes; (void)n_in; (void)out_size; (void)ws_size;
  const float* x      = (const float*)d_in[0];
  const float* w_qkv  = (const float*)d_in[1];
  const float* b_qkv  = (const float*)d_in[2];
  const float* w_proj = (const float*)d_in[3];
  const float* b_proj = (const float*)d_in[4];
  float* out = (float*)d_out;

  unsigned short* ws = (unsigned short*)d_ws;
  const size_t HSZ = (size_t)NHEAD * T_SEQ * HDIM;   // 1.57M elems each
  unsigned short* qw = ws;
  unsigned short* kw = qw + HSZ;
  unsigned short* vw = kw + HSZ;
  unsigned short* yw = vw + HSZ;

  qkv_rope_kernel<<<dim3(36, 32), 128, 0, stream>>>(x, w_qkv, b_qkv, qw, kw, vw);
  attn_kernel<<<dim3(32, 12), 128, 0, stream>>>(qw, kw, vw, yw);
  proj_kernel<<<dim3(12, 32), 128, 0, stream>>>(yw, w_proj, b_proj, out);
}
